// QHeadMultipleActions_47717086659219
// MI455X (gfx1250) — matrix operand order repaired and checked by reading, not compiled, not run on hardware
//
#include <hip/hip_runtime.h>
#include <hip/hip_bf16.h>

// Problem constants (match reference)
#define NBATCH 512
#define SEQ    64
#define DIMV   1024
#define NACT   8
#define NBINS  256
#define DIN    2048           // d_inner = 2*DIM
#define NSTATE 16
#define DTR    64             // dt_rank = ceil(1024/16)
#define XP_N   96             // DTR + 2*NSTATE
#define MTOK   (NBATCH*NACT)  // 4096 tokens

typedef __attribute__((ext_vector_type(16))) __bf16 v16bf;
typedef __attribute__((ext_vector_type(8)))  __bf16 v8bf;
typedef __attribute__((ext_vector_type(8)))  float  v8f;

static __device__ __forceinline__ v16bf cat16(v8bf lo, v8bf hi) {
  return __builtin_shufflevector(lo, hi, 0,1,2,3,4,5,6,7,8,9,10,11,12,13,14,15);
}

// A fragment (16x32 bf16): lane r = row; half h holds K chunks [k+8h..] and [k+16+8h..]
static __device__ __forceinline__ v16bf load_fragA(const __bf16* __restrict__ row, int k, int h) {
  v8bf lo = *reinterpret_cast<const v8bf*>(row + k + 8*h);
  v8bf hi = *reinterpret_cast<const v8bf*>(row + k + 16 + 8*h);
  return cat16(lo, hi);
}
// B fragment (32x16 bf16, KxN): lane r = column; half h holds K chunks [k+8h..] and [k+16+8h..]
static __device__ __forceinline__ v16bf load_fragB(const __bf16* __restrict__ row, int k, int h) {
  v8bf lo = *reinterpret_cast<const v8bf*>(row + k + 8*h);
  v8bf hi = *reinterpret_cast<const v8bf*>(row + k + 16 + 8*h);
  return cat16(lo, hi);
}

// C = A[M,K] * B[N,K]^T. One wave per (MI*16)x(NJ*16) C tile.
// Grid = (N/(NJ*16), M/(MI*16), Z), block = 32.
// __launch_bounds__(32, 1): lift the occupancy-driven VGPR cap so the
// 8-tile accumulator + double-buffered fragments stay in registers.
// EP: 0 = plain f32 store, 1 = sigmoid, 2 = softplus(x + bias[n])
template <int MI, int NJ, int EP>
__global__ void __launch_bounds__(32, 1)
gemm_wmma(const __bf16* __restrict__ Ag, long lda, long sAz,
          const __bf16* __restrict__ Bg, long ldb, long sBz,
          float* __restrict__ Cg, long ldc, long sCz,
          int K, const float* __restrict__ bias)
{
  const int lane = threadIdx.x;
  const int r = lane & 15;
  const int h = lane >> 4;
  const long z = blockIdx.z;
  const __bf16* A  = Ag + z * sAz;
  const __bf16* Bm = Bg + z * sBz;
  float*        C  = Cg + z * sCz;
  const int m0 = blockIdx.y * (MI * 16);
  const int n0 = blockIdx.x * (NJ * 16);

  const __bf16* arow[MI];
  const __bf16* brow[NJ];
#pragma unroll
  for (int i = 0; i < MI; ++i) arow[i] = A  + (long)(m0 + i * 16 + r) * lda;
#pragma unroll
  for (int j = 0; j < NJ; ++j) brow[j] = Bm + (long)(n0 + j * 16 + r) * ldb;

  v8f acc[MI][NJ] = {};
#pragma unroll 2
  for (int k = 0; k < K; k += 32) {
    v16bf va[MI], vb[NJ];
#pragma unroll
    for (int i = 0; i < MI; ++i) va[i] = load_fragA(arow[i], k, h);
#pragma unroll
    for (int j = 0; j < NJ; ++j) vb[j] = load_fragB(brow[j], k, h);
#pragma unroll
    for (int i = 0; i < MI; ++i)
#pragma unroll
      for (int j = 0; j < NJ; ++j)
        acc[i][j] = __builtin_amdgcn_wmma_f32_16x16x32_bf16(
            false, va[i], false, vb[j], (short)0, acc[i][j], false, false);
  }

#pragma unroll
  for (int i = 0; i < MI; ++i) {
#pragma unroll
    for (int j = 0; j < NJ; ++j) {
      const int n = n0 + j * 16 + r;
      const float bval = (EP == 2) ? bias[n] : 0.f;
      float* cp = C + (long)(m0 + i * 16 + 8 * h) * ldc + n;
#pragma unroll
      for (int v = 0; v < 8; ++v) {
        float val = acc[i][j][v];
        if (EP == 2) {                       // softplus(x + bias)
          val += bval;
          val = (val > 20.f) ? val : __logf(1.f + __expf(val));
        } else if (EP == 1) {                // sigmoid
          val = 1.f / (1.f + __expf(-val));
        }
        cp[(long)v * ldc] = val;             // row m0+i*16+8h+v, col n
      }
    }
  }
}

// sos[b,d] = mean over SEQ of encoded_state[b,s,d]
__global__ void mean_sos(const float* __restrict__ es, float* __restrict__ sos) {
  long i = (long)blockIdx.x * blockDim.x + threadIdx.x;
  if (i >= (long)NBATCH * DIMV) return;
  int b = (int)(i >> 10), d = (int)(i & (DIMV - 1));
  const float* p = es + (long)b * SEQ * DIMV + d;
  float acc = 0.f;
  for (int s = 0; s < SEQ; ++s) acc += p[(long)s * DIMV];
  sos[i] = acc * (1.0f / SEQ);
}

// tokens[b,0,:] = sos[b]; tokens[b,l,:] = abe[l-1, actions[b,l-1], :]  (bf16)
__global__ void build_tokens(const float* __restrict__ sos, const float* __restrict__ abe,
                             const int* __restrict__ actions, __bf16* __restrict__ tok) {
  long i = (long)blockIdx.x * blockDim.x + threadIdx.x;
  if (i >= (long)MTOK * DIMV) return;
  int m = (int)(i >> 10), d = (int)(i & (DIMV - 1));
  int b = m >> 3, l = m & 7;
  float v;
  if (l == 0) {
    v = sos[(long)b * DIMV + d];
  } else {
    int idx = l - 1;
    int a = actions[b * (NACT - 1) + idx];
    v = abe[((long)idx * NBINS + a) * DIMV + d];
  }
  tok[i] = (__bf16)v;
}

__global__ void cvt_bf16(const float* __restrict__ in, __bf16* __restrict__ out, long n) {
  long i = (long)blockIdx.x * blockDim.x + threadIdx.x;
  if (i < n) out[i] = (__bf16)in[i];
}

// dt_bf16[m, r] = bf16(x_dbl[m, r]), r < DTR
__global__ void slice_dt(const float* __restrict__ xdbl, __bf16* __restrict__ dt) {
  long i = (long)blockIdx.x * blockDim.x + threadIdx.x;
  if (i >= (long)MTOK * DTR) return;
  int m = (int)(i >> 6), rk = (int)(i & (DTR - 1));
  dt[i] = (__bf16)xdbl[(long)m * XP_N + rk];
}

// causal depthwise conv1d (k=4, left pad) + bias + silu over xh = xz[:, :DIN]
__global__ void conv_silu(const float* __restrict__ xz, const float* __restrict__ w,
                          const float* __restrict__ bias,
                          float* __restrict__ xc, __bf16* __restrict__ xcb) {
  long i = (long)blockIdx.x * blockDim.x + threadIdx.x;
  if (i >= (long)MTOK * DIN) return;
  int m = (int)(i >> 11), c = (int)(i & (DIN - 1));
  int b = m >> 3, l = m & 7;
  float acc = bias[c];
#pragma unroll
  for (int k = 0; k < 4; ++k) {
    int ll = l - 3 + k;
    if (ll >= 0) acc += xz[(long)(b * NACT + ll) * (2 * DIN) + c] * w[c * 4 + k];
  }
  float s = acc / (1.f + __expf(-acc));   // silu
  xc[i] = s;
  xcb[i] = (__bf16)s;
}

// selective scan: block = 256 channels of one batch element; B/C tiles staged in LDS
__global__ void scan_kernel(const float* __restrict__ delta, const float* __restrict__ xc,
                            const float* __restrict__ xdbl, const float* __restrict__ A_log,
                            const float* __restrict__ Dp, const float* __restrict__ xz,
                            __bf16* __restrict__ yb) {
  __shared__ float sB[NACT][NSTATE];
  __shared__ float sC[NACT][NSTATE];
  const int b = blockIdx.y;
  const int c = blockIdx.x * blockDim.x + threadIdx.x;  // channel in [0, DIN)
  const int t = threadIdx.x;
  if (t < NACT * NSTATE) {
    int l = t >> 4, n = t & (NSTATE - 1);
    long base = (long)(b * NACT + l) * XP_N;
    sB[l][n] = xdbl[base + DTR + n];
    sC[l][n] = xdbl[base + DTR + NSTATE + n];
  }
  __syncthreads();

  float Arow[NSTATE];
#pragma unroll
  for (int n = 0; n < NSTATE; ++n) Arow[n] = -__expf(A_log[(long)c * NSTATE + n]);
  const float Dv = Dp[c];

  float hst[NSTATE];
#pragma unroll
  for (int n = 0; n < NSTATE; ++n) hst[n] = 0.f;

  for (int l = 0; l < NACT; ++l) {
    long m = (long)(b * NACT + l);
    float dl = delta[m * DIN + c];
    float u  = xc[m * DIN + c];
    float y  = 0.f;
#pragma unroll
    for (int n = 0; n < NSTATE; ++n) {
      float dA = __expf(dl * Arow[n]);
      hst[n] = dA * hst[n] + dl * sB[l][n] * u;
      y += hst[n] * sC[l][n];
    }
    y += u * Dv;
    float zz = xz[m * (2 * DIN) + DIN + c];
    y *= zz / (1.f + __expf(-zz));        // * silu(z)
    yb[m * DIN + c] = (__bf16)y;
  }
}

// abe rolled by -1 over the bins axis, converted to bf16
__global__ void roll_abe(const float* __restrict__ abe, __bf16* __restrict__ out) {
  long i = (long)blockIdx.x * blockDim.x + threadIdx.x;
  if (i >= (long)NACT * NBINS * DIMV) return;
  int d = (int)(i & (DIMV - 1));
  int a = (int)((i >> 10) & (NBINS - 1));
  int n = (int)(i >> 18);
  out[i] = (__bf16)abe[((long)n * NBINS + ((a + 1) & (NBINS - 1))) * DIMV + d];
}

static inline unsigned blocks_for(long n, int t) { return (unsigned)((n + t - 1) / t); }

extern "C" void kernel_launch(void* const* d_in, const int* in_sizes, int n_in,
                              void* d_out, int out_size, void* d_ws, size_t ws_size,
                              hipStream_t stream) {
  (void)in_sizes; (void)n_in; (void)out_size; (void)ws_size;
  const float* encoded   = (const float*)d_in[0];
  const int*   actions   = (const int*)  d_in[1];
  const float* abe_f     = (const float*)d_in[2];
  /* d_in[3] = gamma (unused by reference) */
  const float* in_proj_w = (const float*)d_in[4];
  const float* conv_w    = (const float*)d_in[5];
  const float* conv_b    = (const float*)d_in[6];
  const float* x_proj_w  = (const float*)d_in[7];
  const float* dt_proj_w = (const float*)d_in[8];
  const float* dt_proj_b = (const float*)d_in[9];
  const float* A_log     = (const float*)d_in[10];
  const float* D_param   = (const float*)d_in[11];
  const float* out_projw = (const float*)d_in[12];
  float* out = (float*)d_out;

  char* ws = (char*)d_ws;
  size_t off = 0;
  auto alloc = [&](size_t bytes) -> char* {
    char* p = ws + off;
    off = (off + bytes + 255) & ~(size_t)255;
    return p;
  };
  float*  sos      = (float*) alloc((size_t)NBATCH * DIMV * 4);
  __bf16* tok_b    = (__bf16*)alloc((size_t)MTOK * DIMV * 2);
  __bf16* w_in_b   = (__bf16*)alloc((size_t)2 * DIN * DIMV * 2);
  float*  xz       = (float*) alloc((size_t)MTOK * 2 * DIN * 4);
  float*  xc       = (float*) alloc((size_t)MTOK * DIN * 4);
  __bf16* xc_b     = (__bf16*)alloc((size_t)MTOK * DIN * 2);
  __bf16* w_xp_b   = (__bf16*)alloc((size_t)XP_N * DIN * 2);
  float*  xdbl     = (float*) alloc((size_t)MTOK * XP_N * 4);
  __bf16* dt_b     = (__bf16*)alloc((size_t)MTOK * DTR * 2);
  __bf16* w_dt_b   = (__bf16*)alloc((size_t)DIN * DTR * 2);
  float*  delta    = (float*) alloc((size_t)MTOK * DIN * 4);
  __bf16* y_b      = (__bf16*)alloc((size_t)MTOK * DIN * 2);
  __bf16* w_out_b  = (__bf16*)alloc((size_t)DIMV * DIN * 2);
  float*  embed    = (float*) alloc((size_t)MTOK * DIMV * 4);
  __bf16* embed_b  = (__bf16*)alloc((size_t)MTOK * DIMV * 2);
  __bf16* abe_b    = (__bf16*)alloc((size_t)NACT * NBINS * DIMV * 2);

  const int T = 256;

  // 1. sos = mean(encoded_state) over seq
  mean_sos<<<blocks_for((long)NBATCH * DIMV, T), T, 0, stream>>>(encoded, sos);
  // 2. tokens (bf16): [sos | gathered bin embeddings]
  build_tokens<<<blocks_for((long)MTOK * DIMV, T), T, 0, stream>>>(sos, abe_f, actions, tok_b);
  // 3. weights -> bf16
  cvt_bf16<<<blocks_for((long)2 * DIN * DIMV, T), T, 0, stream>>>(in_proj_w, w_in_b, (long)2 * DIN * DIMV);
  // 4. GEMM1: xz[4096, 4096] = tokens @ in_proj_w^T   (32x64 tile / wave)
  gemm_wmma<2, 4, 0><<<dim3(2 * DIN / 64, MTOK / 32, 1), 32, 0, stream>>>(
      tok_b, DIMV, 0, w_in_b, DIMV, 0, xz, 2 * DIN, 0, DIMV, nullptr);
  // 5. causal conv + silu on xh = xz[:, :DIN]
  conv_silu<<<blocks_for((long)MTOK * DIN, T), T, 0, stream>>>(xz, conv_w, conv_b, xc, xc_b);
  // 6. GEMM2: x_dbl[4096, 96] = xc @ x_proj_w^T       (32x32 tile / wave)
  cvt_bf16<<<blocks_for((long)XP_N * DIN, T), T, 0, stream>>>(x_proj_w, w_xp_b, (long)XP_N * DIN);
  gemm_wmma<2, 2, 0><<<dim3(XP_N / 32, MTOK / 32, 1), 32, 0, stream>>>(
      xc_b, DIN, 0, w_xp_b, DIN, 0, xdbl, XP_N, 0, DIN, nullptr);
  // 7. GEMM3: delta = softplus(dt @ dt_proj_w^T + dt_proj_b)
  slice_dt<<<blocks_for((long)MTOK * DTR, T), T, 0, stream>>>(xdbl, dt_b);
  cvt_bf16<<<blocks_for((long)DIN * DTR, T), T, 0, stream>>>(dt_proj_w, w_dt_b, (long)DIN * DTR);
  gemm_wmma<2, 4, 2><<<dim3(DIN / 64, MTOK / 32, 1), 32, 0, stream>>>(
      dt_b, DTR, 0, w_dt_b, DTR, 0, delta, DIN, 0, DTR, dt_proj_b);
  // 8. selective scan + D skip + silu(z) gate -> y (bf16)
  scan_kernel<<<dim3(DIN / T, NBATCH), T, 0, stream>>>(delta, xc, xdbl, A_log, D_param, xz, y_b);
  // 9. GEMM4: embed[4096, 1024] = y @ out_proj_w^T    (32x64 tile / wave)
  cvt_bf16<<<blocks_for((long)DIMV * DIN, T), T, 0, stream>>>(out_projw, w_out_b, (long)DIMV * DIN);
  gemm_wmma<2, 4, 0><<<dim3(DIMV / 64, MTOK / 32, 1), 32, 0, stream>>>(
      y_b, DIN, 0, w_out_b, DIN, 0, embed, DIMV, 0, DIN, nullptr);
  // 10. logits: batched over n, sigmoid epilogue straight into d_out
  cvt_bf16<<<blocks_for((long)MTOK * DIMV, T), T, 0, stream>>>(embed, embed_b, (long)MTOK * DIMV);
  roll_abe<<<blocks_for((long)NACT * NBINS * DIMV, T), T, 0, stream>>>(abe_f, abe_b);
  gemm_wmma<2, 4, 1><<<dim3(NBINS / 64, NBATCH / 32, NACT), 32, 0, stream>>>(
      embed_b, (long)NACT * DIMV, DIMV,
      abe_b,   DIMV,              (long)NBINS * DIMV,
      out,     (long)NACT * NBINS, NBINS,
      DIMV, nullptr);
}
